// r_MDGAT2_43430709297720
// MI455X (gfx1250) — compile-verified, matrix operand order checked by reading, not run on hardware
//
#include <hip/hip_runtime.h>
#include <hip/hip_bf16.h>
#include <math.h>

// ---------------------------------------------------------------------------
// MDGAT forward on MI455X (gfx1250).  All matmuls on v_wmma_f32_16x16x32_f16
// (f16 in, f32 accum).  Activations are point-major [point][channel] so every
// WMMA fragment load/store is contiguous (b128 clauses).  Sinkhorn runs fully
// in LDS (264KB of 320KB/WGP).
// ---------------------------------------------------------------------------

typedef __attribute__((ext_vector_type(16))) _Float16 v16h;
typedef __attribute__((ext_vector_type(8)))  float    v8f;

#define NCOL 512   // B*N points
#define NPT  256

// ---- wave32 reductions -----------------------------------------------------
__device__ __forceinline__ float wmaxf(float v) {
    #pragma unroll
    for (int o = 16; o > 0; o >>= 1) v = fmaxf(v, __shfl_xor(v, o, 32));
    return v;
}
__device__ __forceinline__ float wsumf(float v) {
    #pragma unroll
    for (int o = 16; o > 0; o >>= 1) v += __shfl_xor(v, o, 32);
    return v;
}
__device__ __forceinline__ int wsumi(int v) {
    #pragma unroll
    for (int o = 16; o > 0; o >>= 1) v += __shfl_xor(v, o, 32);
    return v;
}

// ===========================================================================
// WMMA GEMM:  Y[n][co] = sum_k W[co][k] * X[n][k] + bias[co]
//   W row-major [M][K] (K multiple of 32), X point-major [512][K],
//   Y point-major [512][ldy].  perm!=0: output row ro is produced by weight
//   row (ro&31)*4 + (ro>>5)  (head-major QKV); bias must be pre-permuted so
//   the epilogue load is contiguous.
// Per the ISA 16-bit A/B fragment map, a lane's 16 elements are the two
// contiguous 8-wide K segments [k0+8g, +7] and [k0+16+8g, +7].
// ===========================================================================
__global__ void gemm_wx(const float* __restrict__ W, const float* __restrict__ X,
                        const float* __restrict__ bias, float* __restrict__ Y,
                        int K, int ldy, int perm) {
    int wid  = blockIdx.x * (blockDim.x >> 5) + (threadIdx.x >> 5);
    int lane = threadIdx.x & 31;
    int tn = wid & 31;        // 32 point tiles (512/16)
    int tm = wid >> 5;        // output-channel tile
    int m16 = lane & 15, g = lane >> 4;
    int ro = tm * 16 + m16;
    int wr = perm ? ((ro & 31) * 4 + (ro >> 5)) : ro;     // one-time, pre-loop
    const float* Wr = W + (size_t)wr * K + 8 * g;
    const float* Xr = X + (size_t)(tn * 16 + m16) * K + 8 * g;
    v8f acc = {};
    for (int k0 = 0; k0 < K; k0 += 32) {
        const float* wa = Wr + k0;
        const float* xb = Xr + k0;
        v16h a, b;
        #pragma unroll
        for (int e = 0; e < 8; ++e) {
            a[e]     = (_Float16)wa[e];
            a[e + 8] = (_Float16)wa[e + 16];
            b[e]     = (_Float16)xb[e];
            b[e + 8] = (_Float16)xb[e + 16];
        }
        acc = __builtin_amdgcn_wmma_f32_16x16x32_f16(false, a, false, b,
                                                     (short)0, acc, false, false);
    }
    float* py = Y + (size_t)(tn * 16 + m16) * ldy + tm * 16 + 8 * g;
    const float* pb = bias + tm * 16 + 8 * g;             // contiguous epilogue
    #pragma unroll
    for (int r = 0; r < 8; ++r) py[r] = acc[r] + pb[r];
}

// Permute a 128-entry bias into head-major output order: Bp[ro]=B[(ro&31)*4+(ro>>5)]
__global__ void perm_bias(const float* __restrict__ B, float* __restrict__ Bp) {
    int ro = threadIdx.x;
    Bp[ro] = B[(ro & 31) * 4 + (ro >> 5)];
}

// ===========================================================================
// Score GEMM: S[(b*nh+h)][n][m] = scale * sum_k Q[b,n][h*hd+k] * K[b,m][h*hd+k]
// Q/K point-major with head-major channels (contiguous fragments).
// ===========================================================================
__global__ void scores_wmma(const float* __restrict__ Q, const float* __restrict__ Km,
                            float* __restrict__ S, int nh, int hd, int ld, float scale) {
    int wid  = blockIdx.x * (blockDim.x >> 5) + (threadIdx.x >> 5);
    int lane = threadIdx.x & 31;
    int tmv = wid & 15;
    int tnv = (wid >> 4) & 15;
    int h   = (wid >> 8) % nh;
    int b   = wid / (256 * nh);
    int m16 = lane & 15, g = lane >> 4;
    const float* Qr = Q  + (size_t)(b * NPT + tnv * 16 + m16) * ld + h * hd + 8 * g;
    const float* Kr = Km + (size_t)(b * NPT + tmv * 16 + m16) * ld + h * hd + 8 * g;
    v8f acc = {};
    for (int k0 = 0; k0 < hd; k0 += 32) {
        v16h a, bb;
        #pragma unroll
        for (int e = 0; e < 8; ++e) {
            a[e]      = (_Float16)Qr[k0 + e];
            a[e + 8]  = (_Float16)Qr[k0 + e + 16];
            bb[e]     = (_Float16)Kr[k0 + e];
            bb[e + 8] = (_Float16)Kr[k0 + e + 16];
        }
        acc = __builtin_amdgcn_wmma_f32_16x16x32_f16(false, a, false, bb,
                                                     (short)0, acc, false, false);
    }
    #pragma unroll
    for (int r = 0; r < 8; ++r) {
        int n = tnv * 16 + r + 8 * g;
        int m = tmv * 16 + m16;
        S[((size_t)(b * nh + h) * NPT + n) * NPT + m] = acc[r] * scale;
    }
}

// ===========================================================================
// Row softmax with optional exact top-k mask (keep = score >= kth-largest).
// One wave per 256-wide row; 8 contiguous values per lane (b128 loads);
// exact k-th largest via binary search in order-preserving uint keyspace.
// ===========================================================================
__global__ void softmax_topk(float* __restrict__ S, int topk) {
    int wave = threadIdx.x >> 5, lane = threadIdx.x & 31;
    size_t row = (size_t)blockIdx.x * 8 + wave;
    float* r = S + row * NPT;
    float vals[8]; unsigned keys[8]; bool mask[8];
    #pragma unroll
    for (int c = 0; c < 8; ++c) {
        vals[c] = r[lane * 8 + c];
        unsigned f = __float_as_uint(vals[c]);
        keys[c] = f ^ ((f >> 31) ? 0xFFFFFFFFu : 0x80000000u);
        mask[c] = true;
    }
    if (topk > 0 && topk < NPT) {
        unsigned lo = 0u, hi = 0xFFFFFFFFu;
        for (int it = 0; it < 33 && lo < hi; ++it) {
            unsigned mid = lo + ((hi - lo) >> 1) + ((hi - lo) & 1u);
            int cnt = 0;
            #pragma unroll
            for (int c = 0; c < 8; ++c) cnt += (keys[c] >= mid) ? 1 : 0;
            cnt = wsumi(cnt);
            if (cnt >= topk) lo = mid; else hi = mid - 1u;
        }
        #pragma unroll
        for (int c = 0; c < 8; ++c) mask[c] = (keys[c] >= lo);
    }
    float mx = -INFINITY;
    #pragma unroll
    for (int c = 0; c < 8; ++c) if (mask[c]) mx = fmaxf(mx, vals[c]);
    mx = wmaxf(mx);
    float ex[8], sm = 0.f;
    #pragma unroll
    for (int c = 0; c < 8; ++c) { ex[c] = mask[c] ? __expf(vals[c] - mx) : 0.f; sm += ex[c]; }
    sm = wsumf(sm);
    float inv = 1.f / sm;
    #pragma unroll
    for (int c = 0; c < 8; ++c) r[lane * 8 + c] = ex[c] * inv;
}

// Transpose [512][128] point-major -> [128][512] channel-major (for apply B).
__global__ void transpose_cm(const float* __restrict__ X, float* __restrict__ Xt) {
    int idx = blockIdx.x * blockDim.x + threadIdx.x;   // 65536
    int bn = idx >> 7, c = idx & 127;
    Xt[(size_t)c * NCOL + bn] = X[idx];
}

// ===========================================================================
// Cross attention apply: out[n][d*4+h] = sum_m P[b,h,n,m] * Vt[h*32+d][b,m]
// A (P rows) and B (Vt rows) fragments are both contiguous.
// ===========================================================================
__global__ void apply_cross(const float* __restrict__ P, const float* __restrict__ Vt,
                            float* __restrict__ O) {
    int wid  = blockIdx.x * (blockDim.x >> 5) + (threadIdx.x >> 5);
    int lane = threadIdx.x & 31;
    int tn = wid & 15, dt = (wid >> 4) & 1, h = (wid >> 5) & 3, b = (wid >> 7) & 1;
    int m16 = lane & 15, g = lane >> 4;
    const float* Pr = P + (size_t)(b * 4 + h) * (NPT * NPT)
                        + (size_t)(tn * 16 + m16) * NPT + 8 * g;
    int d = dt * 16 + m16;
    const float* Vr = Vt + (size_t)(h * 32 + d) * NCOL + b * NPT + 8 * g;
    v8f acc = {};
    for (int k0 = 0; k0 < NPT; k0 += 32) {
        v16h a, bb;
        #pragma unroll
        for (int e = 0; e < 8; ++e) {
            a[e]      = (_Float16)Pr[k0 + e];
            a[e + 8]  = (_Float16)Pr[k0 + e + 16];
            bb[e]     = (_Float16)Vr[k0 + e];
            bb[e + 8] = (_Float16)Vr[k0 + e + 16];
        }
        acc = __builtin_amdgcn_wmma_f32_16x16x32_f16(false, a, false, bb,
                                                     (short)0, acc, false, false);
    }
    int dd = dt * 16 + (lane & 15);
    #pragma unroll
    for (int r = 0; r < 8; ++r) {
        int n = tn * 16 + r + 8 * g;
        O[(size_t)(b * NPT + n) * 128 + dd * 4 + h] = acc[r];   // interleaved order
    }
}

// ===========================================================================
// Self attention apply with rotary value modulation:
//   out[d] = (P o cos th(m,n)) @ V_sel + (P o sin th(m,n)) @ Vrot_sel
// tt: 0 -> alpha (d%4 in {0,1}), 1 -> phi (d%4 in {2,3}); those 16 channels per
// head are exactly one WMMA tile width.  Trig stored [n][m] => contiguous.
// ===========================================================================
__global__ void apply_self(const float* __restrict__ P, const float* __restrict__ Vt,
                           const float* __restrict__ TRIGs, float* __restrict__ O) {
    int wid  = blockIdx.x * (blockDim.x >> 5) + (threadIdx.x >> 5);
    int lane = threadIdx.x & 31;
    int tn = wid & 15, tt = (wid >> 4) & 1, h = (wid >> 5) & 3, b = (wid >> 7) & 1;
    int m16 = lane & 15, g = lane >> 4;
    int arow = tn * 16 + m16;
    const float* T  = TRIGs + (size_t)b * 4 * (NPT * NPT);
    const float* Tc = T + (size_t)(tt * 2) * (NPT * NPT) + (size_t)arow * NPT + 8 * g;
    const float* Ts = T + (size_t)(tt * 2 + 1) * (NPT * NPT) + (size_t)arow * NPT + 8 * g;
    const float* Pr = P + (size_t)(b * 4 + h) * (NPT * NPT) + (size_t)arow * NPT + 8 * g;
    int col = m16, q = col >> 1, w = col & 1;
    int d  = q * 4 + tt * 2 + w;               // per-head dim, d%4 in angle pair
    int d2 = (w == 0) ? d + 1 : d - 1;         // rotate-half partner
    float sgn = (w == 0) ? -1.f : 1.f;
    const float* Vr  = Vt + (size_t)(h * 32 + d)  * NCOL + b * NPT + 8 * g;
    const float* Vr2 = Vt + (size_t)(h * 32 + d2) * NCOL + b * NPT + 8 * g;
    v8f acc = {};
    for (int k0 = 0; k0 < NPT; k0 += 32) {
        v16h ac, as, bv, bd;
        #pragma unroll
        for (int e = 0; e < 8; ++e) {
            float p0 = Pr[k0 + e], p1 = Pr[k0 + e + 16];
            ac[e]     = (_Float16)(p0 * Tc[k0 + e]);
            ac[e + 8] = (_Float16)(p1 * Tc[k0 + e + 16]);
            as[e]     = (_Float16)(p0 * Ts[k0 + e]);
            as[e + 8] = (_Float16)(p1 * Ts[k0 + e + 16]);
            bv[e]     = (_Float16)Vr[k0 + e];
            bv[e + 8] = (_Float16)Vr[k0 + e + 16];
            bd[e]     = (_Float16)(sgn * Vr2[k0 + e]);
            bd[e + 8] = (_Float16)(sgn * Vr2[k0 + e + 16]);
        }
        acc = __builtin_amdgcn_wmma_f32_16x16x32_f16(false, ac, false, bv,
                                                     (short)0, acc, false, false);
        acc = __builtin_amdgcn_wmma_f32_16x16x32_f16(false, as, false, bd,
                                                     (short)0, acc, false, false);
    }
    #pragma unroll
    for (int r = 0; r < 8; ++r) {
        int n = tn * 16 + r + 8 * g;
        O[(size_t)(b * NPT + n) * 128 + d * 4 + h] = acc[r];    // interleaved order
    }
}

// ===========================================================================
// Rotary angle tables, stored T[t][n][m] = {cos,sin}{alpha,phi}(m,n) with
// p = lrf[m]^T (kpts[m]-kpts[n]).  blockIdx=n, thread=m => contiguous writes.
// ===========================================================================
__global__ void build_trig(const float* __restrict__ kp, const float* __restrict__ lrf,
                           float* __restrict__ T) {
    int n = blockIdx.x, m = threadIdx.x;
    float rx = kp[m * 3 + 0] - kp[n * 3 + 0];
    float ry = kp[m * 3 + 1] - kp[n * 3 + 1];
    float rz = kp[m * 3 + 2] - kp[n * 3 + 2];
    const float* L = lrf + (size_t)m * 9;              // L[y*3+x]
    float px = L[0] * rx + L[3] * ry + L[6] * rz;
    float py = L[1] * rx + L[4] * ry + L[7] * rz;
    float pz = L[2] * rx + L[5] * ry + L[8] * rz;
    float alpha = atanf(py / px);
    if (px < 0.f) alpha += 3.14159265358979323846f;
    float phi = atanf(pz / sqrtf(px * px + py * py));
    if (alpha != alpha) alpha = 0.f;
    if (phi != phi)     phi = 0.f;
    int o = n * NPT + m;
    T[0 * NPT * NPT + o] = __cosf(alpha);
    T[1 * NPT * NPT + o] = __sinf(alpha);
    T[2 * NPT * NPT + o] = __cosf(phi);
    T[3 * NPT * NPT + o] = __sinf(phi);
}

// ===========================================================================
// BatchNorm (training-mode batch stats over all 512 points) + fused ReLU.
// Point-major X[512][C].
// ===========================================================================
__global__ void bn_stats(const float* __restrict__ X, float* __restrict__ mu,
                         float* __restrict__ rinv, int C) {
    int c = blockIdx.x, t = threadIdx.x;
    float s = 0.f, s2 = 0.f;
    for (int j = t; j < NCOL; j += 256) {
        float x = X[(size_t)j * C + c];
        s += x; s2 += x * x;
    }
    __shared__ float sh[256], sh2[256];
    sh[t] = s; sh2[t] = s2; __syncthreads();
    for (int o = 128; o > 0; o >>= 1) {
        if (t < o) { sh[t] += sh[t + o]; sh2[t] += sh2[t + o]; }
        __syncthreads();
    }
    if (t == 0) {
        float m = sh[0] / (float)NCOL;
        float var = sh2[0] / (float)NCOL - m * m;       // biased var (ddof=0)
        mu[c] = m; rinv[c] = rsqrtf(var + 1e-5f);
    }
}

__global__ void bn_relu(float* __restrict__ X, const float* __restrict__ mu,
                        const float* __restrict__ rinv, const float* __restrict__ gamma,
                        const float* __restrict__ beta, int Cmask) {
    int idx = blockIdx.x * blockDim.x + threadIdx.x;
    int c = idx & Cmask;                                 // C is a power of two
    float x = (X[idx] - mu[c]) * rinv[c] * gamma[c] + beta[c];
    X[idx] = fmaxf(x, 0.f);
}

__global__ void vadd(float* __restrict__ d, const float* __restrict__ a,
                     const float* __restrict__ b, int n) {
    int i = blockIdx.x * blockDim.x + threadIdx.x;
    if (i < n) d[i] = a[i] + b[i];
}

// CAT[bn][0:128]=A[bn][:], CAT[bn][128:256]=B[bn][:]
__global__ void concat2(const float* __restrict__ A, const float* __restrict__ B,
                        float* __restrict__ C_) {
    int idx = blockIdx.x * blockDim.x + threadIdx.x;    // 512*256
    int bn = idx >> 8, c = idx & 255;
    C_[idx] = (c < 128) ? A[(size_t)bn * 128 + c] : B[(size_t)bn * 128 + c - 128];
}

// Zero-pad weight matrix W[M][K] -> Wp[M][Kp]
__global__ void pad_w(const float* __restrict__ W, float* __restrict__ Wp,
                      int M, int K, int Kp) {
    int idx = blockIdx.x * blockDim.x + threadIdx.x;
    if (idx >= M * Kp) return;
    int m = idx / Kp, k = idx % Kp;
    Wp[idx] = (k < K) ? W[(size_t)m * K + k] : 0.f;
}

// Encoder inputs, point-major, zero-padded: kin[512][32], din[512][64]
__global__ void pack_enc(const float* __restrict__ kpts, const float* __restrict__ sc,
                         const float* __restrict__ fpfh, float* __restrict__ kin,
                         float* __restrict__ din) {
    int idx = blockIdx.x * blockDim.x + threadIdx.x;
    if (idx >= NCOL) return;
    float* kr = kin + (size_t)idx * 32;
    for (int c = 0; c < 3; ++c) kr[c] = kpts[(size_t)idx * 3 + c];
    kr[3] = sc[idx];
    for (int c = 4; c < 32; ++c) kr[c] = 0.f;
    float* dr = din + (size_t)idx * 64;
    for (int c = 0; c < 33; ++c) dr[c] = fpfh[(size_t)idx * 33 + c];
    for (int c = 33; c < 64; ++c) dr[c] = 0.f;
}

// ===========================================================================
// Log-domain Sinkhorn: one workgroup per batch; 257x257 couplings in LDS
// (264KB of the 320KB WGP LDS), 100 row/col LSE sweeps fully on-chip.
// ===========================================================================
__global__ __launch_bounds__(1024) void sinkhorn(const float* __restrict__ sc,
                                                 const float* __restrict__ alpha_p,
                                                 float* __restrict__ out) {
    extern __shared__ float smem[];
    const int M1 = NPT + 1;                 // 257
    float* coup = smem;                     // 257*257
    float* u = smem + M1 * M1;              // 257
    float* v = u + M1;                      // 257
    const int b = blockIdx.x;
    const float alpha = alpha_p[0];
    for (int idx = threadIdx.x; idx < M1 * M1; idx += blockDim.x) {
        int i = idx / M1, j = idx % M1;
        coup[idx] = (i < NPT && j < NPT) ? sc[((size_t)b * NPT + i) * NPT + j] : alpha;
    }
    for (int i = threadIdx.x; i < M1; i += blockDim.x) { u[i] = 0.f; v[i] = 0.f; }
    __syncthreads();
    const float norm    = -logf(2.f * NPT);
    const float lm_last = logf((float)NPT) + norm;
    int wave = threadIdx.x >> 5, lane = threadIdx.x & 31;
    for (int it = 0; it < 100; ++it) {
        for (int i = wave; i < M1; i += 32) {               // u update (rows)
            float mx = -INFINITY;
            for (int j = lane; j < M1; j += 32) mx = fmaxf(mx, coup[i * M1 + j] + v[j]);
            mx = wmaxf(mx);
            float sm = 0.f;
            for (int j = lane; j < M1; j += 32) sm += __expf(coup[i * M1 + j] + v[j] - mx);
            sm = wsumf(sm);
            if (lane == 0) u[i] = ((i < NPT) ? norm : lm_last) - (mx + logf(sm));
        }
        __syncthreads();
        for (int j = wave; j < M1; j += 32) {               // v update (cols)
            float mx = -INFINITY;
            for (int i = lane; i < M1; i += 32) mx = fmaxf(mx, coup[i * M1 + j] + u[i]);
            mx = wmaxf(mx);
            float sm = 0.f;
            for (int i = lane; i < M1; i += 32) sm += __expf(coup[i * M1 + j] + u[i] - mx);
            sm = wsumf(sm);
            if (lane == 0) v[j] = ((j < NPT) ? norm : lm_last) - (mx + logf(sm));
        }
        __syncthreads();
    }
    for (int idx = threadIdx.x; idx < M1 * M1; idx += blockDim.x) {
        int i = idx / M1, j = idx % M1;
        out[(size_t)b * M1 * M1 + idx] = coup[idx] + u[i] + v[j] - norm;
    }
}

// ===========================================================================
// Host orchestration
// ===========================================================================
// Param leaf indices assuming pytree flatten with alphabetical dict keys:
//  8: bin_score
//  9..18:  denc  (l0:{b,beta,gamma,w} l1:{b,beta,gamma,w} l2:{b,w})
// 19..20:  final {b,w}
// 21..104: gnn x6, 14 leaves each:
//          k.b,k.w,merge.b,merge.w,q.b,q.w,v.b,v.w,
//          mlp0.b,mlp0.beta,mlp0.gamma,mlp0.w,mlp1.b,mlp1.w
// 105..118: kenc (l0..l2:{b,beta,gamma,w}, l3:{b,w})

extern "C" void kernel_launch(void* const* d_in, const int* in_sizes, int n_in,
                              void* d_out, int out_size, void* d_ws, size_t ws_size,
                              hipStream_t stream) {
    (void)in_sizes; (void)out_size; (void)ws_size;
    if (n_in < 119) return;
    auto pf = [&](int i) { return (const float*)d_in[i]; };

    // ---- workspace carve-up (floats) ----
    float* Wbase = (float*)d_ws;
    size_t off = 0;
    auto alloc = [&](size_t n) { float* p = Wbase + off; off += n; return p; };
    float* KIN   = alloc((size_t)NCOL * 32);
    float* DIN   = alloc((size_t)NCOL * 64);
    float* WKP   = alloc(32 * 32);                     // kenc0 padded weights
    float* WDP   = alloc(64 * 64);                     // denc0 padded weights
    float* ACT_A = alloc((size_t)NCOL * 256);
    float* ACT_B = alloc((size_t)NCOL * 256);
    float* KOUT  = alloc((size_t)NCOL * 128);
    float* DESC[2] = { alloc((size_t)NCOL * 128), alloc((size_t)NCOL * 128) };
    float* Qb    = alloc((size_t)NCOL * 128);
    float* Kb    = alloc((size_t)NCOL * 128);
    float* Vb    = alloc((size_t)NCOL * 128);
    float* VT    = alloc((size_t)128 * NCOL);
    float* ATT   = alloc((size_t)NCOL * 128);
    float* MSG   = alloc((size_t)NCOL * 128);
    float* CAT   = alloc((size_t)NCOL * 256);
    float* HID   = alloc((size_t)NCOL * 256);
    float* DT[2] = { alloc((size_t)NCOL * 128), alloc((size_t)NCOL * 128) };
    float* MD0   = alloc((size_t)NCOL * 128);
    float* MD1   = alloc((size_t)NCOL * 128);
    float* P     = alloc((size_t)2 * 4 * NPT * NPT);
    float* TRIG  = alloc((size_t)2 * 2 * 4 * NPT * NPT);
    float* SC    = alloc((size_t)2 * NPT * NPT);
    float* MU    = alloc(256);
    float* RINV  = alloc(256);
    float* BQP   = alloc(128);                         // permuted QKV biases
    float* BKP   = alloc(128);
    float* BVP   = alloc(128);

    auto gemmL = [&](const float* Wp, const float* X, const float* bias, float* Y,
                     int M, int K, int perm) {
        int tiles = (M / 16) * 32;
        gemm_wx<<<tiles / 4, 128, 0, stream>>>(Wp, X, bias, Y, K, M, perm);
    };
    auto bnL = [&](float* X, const float* gamma, const float* beta, int C) {
        bn_stats<<<C, 256, 0, stream>>>(X, MU, RINV, C);
        bn_relu<<<(C * NCOL) / 256, 256, 0, stream>>>(X, MU, RINV, gamma, beta, C - 1);
    };

    // ---- pad odd-K weights once per call ----
    pad_w<<<(32 * 32) / 256, 256, 0, stream>>>(pf(108), WKP, 32, 4, 32);
    pad_w<<<(64 * 64) / 256, 256, 0, stream>>>(pf(12),  WDP, 64, 33, 64);

    // ---------------- encoders (per stream; BN stats are per-call) ----------
    for (int s = 0; s < 2; ++s) {
        pack_enc<<<2, 256, 0, stream>>>(pf(0 + s), pf(2 + s), pf(4 + s), KIN, DIN);
        // kenc: 4->32->64->128->128
        gemmL(WKP,     KIN,   pf(105), ACT_A, 32, 32, 0);  bnL(ACT_A, pf(107), pf(106), 32);
        gemmL(pf(112), ACT_A, pf(109), ACT_B, 64, 32, 0);  bnL(ACT_B, pf(111), pf(110), 64);
        gemmL(pf(116), ACT_B, pf(113), ACT_A, 128, 64, 0); bnL(ACT_A, pf(115), pf(114), 128);
        gemmL(pf(118), ACT_A, pf(117), KOUT, 128, 128, 0);
        // denc: 33->64->128->128
        gemmL(WDP,     DIN,   pf(9),  ACT_A, 64, 64, 0);   bnL(ACT_A, pf(11), pf(10), 64);
        gemmL(pf(16),  ACT_A, pf(13), ACT_B, 128, 64, 0);  bnL(ACT_B, pf(15), pf(14), 128);
        gemmL(pf(18),  ACT_B, pf(17), ACT_A, 128, 128, 0);
        vadd<<<(128 * NCOL) / 256, 256, 0, stream>>>(DESC[s], ACT_A, KOUT, 128 * NCOL);
    }

    // ---------------- rotary trig tables (once, reused by 3 self layers) ----
    for (int s = 0; s < 2; ++s)
        for (int b = 0; b < 2; ++b)
            build_trig<<<256, 256, 0, stream>>>(pf(0 + s) + (size_t)b * NPT * 3,
                                                pf(6 + s) + (size_t)b * NPT * 9,
                                                TRIG + ((size_t)(s * 2 + b) * 4) * NPT * NPT);

    // ---------------- GNN layers -------------------------------------------
    const int kvals[6] = { 0, 0, 128, 128, 64, 64 };
    const float inv_sqrt_hd = 0.17677669529663687f;     // 1/sqrt(32)
    for (int L = 0; L < 6; ++L) {
        int base = 21 + L * 14;
        bool is_self = (L % 2 == 0);
        int topk = (is_self && kvals[L] > 0) ? kvals[L] : 0;
        // permute QKV biases into head-major order (contiguous GEMM epilogue)
        perm_bias<<<1, 128, 0, stream>>>(pf(base + 4), BQP);
        perm_bias<<<1, 128, 0, stream>>>(pf(base + 0), BKP);
        perm_bias<<<1, 128, 0, stream>>>(pf(base + 6), BVP);
        for (int s = 0; s < 2; ++s) {
            const float* src = is_self ? DESC[s] : DESC[1 - s];
            gemmL(pf(base + 5), DESC[s], BQP, Qb, 128, 128, 1);   // q (head-major)
            gemmL(pf(base + 1), src,     BKP, Kb, 128, 128, 1);   // k (head-major)
            gemmL(pf(base + 7), src,     BVP, Vb, 128, 128, 1);   // v (head-major)
            transpose_cm<<<(128 * NCOL) / 256, 256, 0, stream>>>(Vb, VT);
            scores_wmma<<<512, 128, 0, stream>>>(Qb, Kb, P, 4, 32, 128, inv_sqrt_hd);
            softmax_topk<<<256, 256, 0, stream>>>(P, topk);
            if (is_self)
                apply_self<<<64, 128, 0, stream>>>(P, VT,
                        TRIG + (size_t)s * 2 * 4 * NPT * NPT, ATT);
            else
                apply_cross<<<64, 128, 0, stream>>>(P, VT, ATT);
            gemmL(pf(base + 3), ATT, pf(base + 2), MSG, 128, 128, 0);     // merge
            concat2<<<(256 * NCOL) / 256, 256, 0, stream>>>(DESC[s], MSG, CAT);
            gemmL(pf(base + 11), CAT, pf(base + 8), HID, 256, 256, 0);    // mlp0
            bnL(HID, pf(base + 10), pf(base + 9), 256);
            gemmL(pf(base + 13), HID, pf(base + 12), DT[s], 128, 256, 0); // mlp1
        }
        vadd<<<(128 * NCOL) / 256, 256, 0, stream>>>(DESC[0], DESC[0], DT[0], 128 * NCOL);
        vadd<<<(128 * NCOL) / 256, 256, 0, stream>>>(DESC[1], DESC[1], DT[1], 128 * NCOL);
    }

    // ---------------- final projection + correlation -----------------------
    gemmL(pf(20), DESC[0], pf(19), MD0, 128, 128, 0);
    gemmL(pf(20), DESC[1], pf(19), MD1, 128, 128, 0);
    scores_wmma<<<128, 128, 0, stream>>>(MD0, MD1, SC, 1, 128, 128,
                                         0.08838834764831845f);           // 1/sqrt(128)

    // ---------------- Sinkhorn in LDS --------------------------------------
    size_t smem = ((size_t)257 * 257 + 2 * 257 + 2) * sizeof(float);      // ~266KB
    sinkhorn<<<2, 1024, smem, stream>>>(SC, pf(8), (float*)d_out);
}
